// durationLMAdapt_79559974191183
// MI455X (gfx1250) — matrix operand order repaired and checked by reading, not compiled, not run on hardware
//
#include <hip/hip_runtime.h>
#include <hip/hip_bf16.h>
#include <math.h>

// ---------------------------------------------------------------------------
// durationLMAdapt on MI455X (gfx1250, wave32, WMMA bf16 16x16x32)
//   B=64, S=512, D=H=256, NH=4, NH*H=1024
// Phases:
//   1) weights -> bf16, bias fuse
//   2) xproj = gather(emb) @ Wi^T + (bi+bh)   (hoisted out of the scan; WMMA)
//   3) serial gated recurrence: 4 persistent WGs, per-step WMMA GEMM,
//      f32 h-state in registers, bf16 A-operand copy in LDS,
//      software-pipelined xproj/fix_src loads + prefetch
//   4) final FCs (WMMA + tiny VALU dot)
// ---------------------------------------------------------------------------

typedef __attribute__((ext_vector_type(16))) __bf16 v16bf;
typedef __attribute__((ext_vector_type(8)))  __bf16 v8bf;
typedef __attribute__((ext_vector_type(8)))  float  v8f;
typedef __attribute__((ext_vector_type(4)))  float  v4f;

#define B_   64
#define S_   512
#define D_   256
#define H_   256
#define NH_  4
#define NHH_ 1024

__device__ __forceinline__ v16bf join16(v8bf lo, v8bf hi) {
  union { v16bf v; v8bf h[2]; } u;
  u.h[0] = lo; u.h[1] = hi;
  return u.v;
}

__device__ __forceinline__ v8bf cvt8(const float* p) {
  const v4f* q = (const v4f*)p;
  v4f a = q[0], c = q[1];
  v8bf r;
  r[0] = (__bf16)a[0]; r[1] = (__bf16)a[1]; r[2] = (__bf16)a[2]; r[3] = (__bf16)a[3];
  r[4] = (__bf16)c[0]; r[5] = (__bf16)c[1]; r[6] = (__bf16)c[2]; r[7] = (__bf16)c[3];
  return r;
}

__device__ __forceinline__ v8f wmma_bf16(v16bf a, v16bf b, v8f c) {
  return __builtin_amdgcn_wmma_f32_16x16x32_bf16(
      false, a, false, b, (short)0, c, false, false);
}

__device__ __forceinline__ float sigmoidf_(float x) {
  return 1.0f / (1.0f + __expf(-x));
}

// ---------------------------------------------------------------------------
__global__ void convert_weights(const float* __restrict__ Wi,
                                const float* __restrict__ Wh,
                                const float* __restrict__ fc1W,
                                const float* __restrict__ bi,
                                const float* __restrict__ bh,
                                __bf16* __restrict__ Wi16,
                                __bf16* __restrict__ Wh16,
                                __bf16* __restrict__ fc116,
                                float*  __restrict__ bvec) {
  int tid = blockIdx.x * blockDim.x + threadIdx.x;
  int stride = gridDim.x * blockDim.x;
  for (int i = tid; i < H_ * D_;   i += stride) Wi16[i]  = (__bf16)Wi[i];
  for (int i = tid; i < H_ * NHH_; i += stride) Wh16[i]  = (__bf16)Wh[i];
  for (int i = tid; i < D_ * NHH_; i += stride) fc116[i] = (__bf16)fc1W[i];
  for (int i = tid; i < H_;        i += stride) bvec[i]  = bi[i] + bh[i];
}

// ---------------------------------------------------------------------------
// xproj[m,:] = emb_table[src[b,t]] @ Wi^T + (bi+bh),  m = t*64 + b
// M=32768, N=256, K=256. One wave per 16x16 output tile.
// A layout (ISA 7.12.2): lanes 0-15 hold rows, K in {kc..+7, kc+16..+23};
// lanes 16-31 same rows, K in {kc+8..+15, kc+24..+31}. B lane n = column n
// of Wi^T = contiguous row n of row-major Wi.
// ---------------------------------------------------------------------------
__global__ __launch_bounds__(256) void xproj_gemm(const int*    __restrict__ src,
                                                  const float*  __restrict__ emb,
                                                  const __bf16* __restrict__ Wi16,
                                                  const float*  __restrict__ bvec,
                                                  float*        __restrict__ xproj) {
  const int lane  = threadIdx.x & 31;
  const int wave  = threadIdx.x >> 5;
  const int tile  = blockIdx.x * 8 + wave;   // 0..32767
  const int mtile = tile >> 4;
  const int nt    = tile & 15;
  const int m0    = mtile * 16;
  const int khalf = lane >> 4;
  const int colg  = nt * 16 + (lane & 15);

  const int rowA = m0 + (lane & 15);
  const int t = rowA >> 6, b = rowA & 63;
  const float*  arow  = emb  + (size_t)src[b * S_ + t] * D_;
  const __bf16* bbase = Wi16 + (size_t)colg * D_;

  v8f acc = {};
#pragma unroll
  for (int kc = 0; kc < D_; kc += 32) {
    v8bf a0 = cvt8(arow + kc + khalf * 8);
    v8bf a1 = cvt8(arow + kc + 16 + khalf * 8);
    const v8bf* bp = (const v8bf*)(bbase + kc + khalf * 8);
    acc = wmma_bf16(join16(a0, a1), join16(bp[0], bp[2]), acc);
  }
  const float bias = bvec[colg];
#pragma unroll
  for (int r = 0; r < 8; ++r) {
    int rowm = m0 + r + khalf * 8;
    xproj[(size_t)rowm * H_ + colg] = acc[r] + bias;
  }
}

// ---------------------------------------------------------------------------
// Serial recurrence. 4 WGs x 512 threads (16 waves). Each WG owns 16 batch
// rows. Per step: [16,1024] x [1024,256] bf16 WMMA GEMM (A = h in LDS bf16,
// B = Wh from L2), gate epilogue against f32 h-state held in REGISTERS
// (each lane owns exactly its 8 rows x 4 channels at its column), barrier.
// xproj / fix_src for step t+1 are loaded during step t's GEMM.
// ---------------------------------------------------------------------------
__global__ __launch_bounds__(512) void recurrence(const int*    __restrict__ input_len,
                                                  const float*  __restrict__ fix_src,
                                                  const __bf16* __restrict__ Wh16,
                                                  const float*  __restrict__ xproj,
                                                  __bf16*       __restrict__ sel16) {
  __shared__ __bf16 hcat16[16][NHH_];   // 32 KB : WMMA A operand (bf16 h)
  __shared__ int    lenm1[16];

  const int b0  = blockIdx.x * 16;
  const int tid = threadIdx.x;

  for (int i = tid; i < 16 * NHH_; i += 512)
    ((__bf16*)hcat16)[i] = (__bf16)0.0f;
  if (tid < 16) lenm1[tid] = input_len[b0 + tid] - 1;
  __syncthreads();

  const int lane  = tid & 31;
  const int nt    = tid >> 5;            // wave = N tile (16 waves -> 256 cols)
  const int khalf = lane >> 4;
  const int colg  = nt * 16 + (lane & 15);
  const int arow  = lane & 15;
  const __bf16* bbase = Wh16 + (size_t)colg * NHH_;

  // f32 hidden state in registers: hreg[r][nh] = h[nh, b0+row, colg]
  float hreg[8][NH_];
#pragma unroll
  for (int r = 0; r < 8; ++r)
#pragma unroll
    for (int nh = 0; nh < NH_; ++nh) hreg[r][nh] = 0.0f;

  // prologue: load step-0 inputs
  float xp_cur[8], d_cur[8];
#pragma unroll
  for (int r = 0; r < 8; ++r) {
    int row = r + khalf * 8;
    xp_cur[r] = xproj[(size_t)(b0 + row) * H_ + colg];          // t = 0
    d_cur[r]  = fix_src[(size_t)(b0 + row) * S_];               // t = 0
  }

  for (int t = 0; t < S_; ++t) {
    v8f acc = {};
#pragma unroll 4
    for (int kc = 0; kc < NHH_; kc += 32) {
      const v8bf* a0 = (const v8bf*)&hcat16[arow][kc + khalf * 8];
      const v8bf* a1 = (const v8bf*)&hcat16[arow][kc + 16 + khalf * 8];
      const v8bf* bp = (const v8bf*)(bbase + kc + khalf * 8);
      acc = wmma_bf16(join16(a0[0], a1[0]), join16(bp[0], bp[2]), acc);
    }

    // software-pipelined loads for step t+1 (overlap with WMMA latency)
    float xp_nxt[8], d_nxt[8];
    const int tn = (t + 1 < S_) ? (t + 1) : t;
#pragma unroll
    for (int r = 0; r < 8; ++r) {
      int row = r + khalf * 8;
      xp_nxt[r] = xproj[(size_t)(tn * B_ + b0 + row) * H_ + colg];
      d_nxt[r]  = fix_src[(size_t)(b0 + row) * S_ + tn];
    }
    if (t + 2 < S_) {   // warm L2/WGP$ for t+2 (global_prefetch_b8)
      __builtin_prefetch(&xproj[(size_t)((t + 2) * B_ + b0 + khalf * 8) * H_ + colg], 0, 0);
    }

    float cand[8];
#pragma unroll
    for (int r = 0; r < 8; ++r)
      cand[r] = tanhf(acc[r] + xp_cur[r]);

    __syncthreads();  // all WMMA reads of hcat16 complete for this step

#pragma unroll
    for (int r = 0; r < 8; ++r) {
      int row = r + khalf * 8;
#pragma unroll
      for (int nh = 0; nh < NH_; ++nh) {
        float g  = sigmoidf_(d_cur[r] - 3.0f * nh);   // SCALE_RATE=1, MAX_DUR/NH=3
        float hn = g * cand[r] + (1.0f - g) * hreg[r][nh];
        hreg[r][nh] = hn;
        hcat16[row][nh * H_ + colg] = (__bf16)hn;
      }
    }
    __syncthreads();  // updated state visible

    // capture rows whose last valid step is t (uniform branch: lenm1 in LDS)
    for (int row = 0; row < 16; ++row) {
      if (lenm1[row] == t) {
        for (int j = tid; j < NHH_; j += 512)
          sel16[(size_t)(b0 + row) * NHH_ + j] = hcat16[row][j];
      }
    }

#pragma unroll
    for (int r = 0; r < 8; ++r) { xp_cur[r] = xp_nxt[r]; d_cur[r] = d_nxt[r]; }
  }
}

// ---------------------------------------------------------------------------
// hid = tanh(sel @ fc1^T + b1), out = hid @ fc2^T + b2
// ---------------------------------------------------------------------------
__global__ __launch_bounds__(512) void final_fc(const __bf16* __restrict__ sel16,
                                                const __bf16* __restrict__ fc116,
                                                const float*  __restrict__ fc1b,
                                                const float*  __restrict__ fc2W,
                                                const float*  __restrict__ fc2b,
                                                float*        __restrict__ out) {
  __shared__ float hid[B_][H_];   // 64 KB
  const int tid   = threadIdx.x;
  const int lane  = tid & 31;
  const int nt    = tid >> 5;
  const int khalf = lane >> 4;
  const int colg  = nt * 16 + (lane & 15);
  const __bf16* bbase = fc116 + (size_t)colg * NHH_;

  for (int mt = 0; mt < 4; ++mt) {
    const int m0 = mt * 16;
    const __bf16* abase = sel16 + (size_t)(m0 + (lane & 15)) * NHH_;
    v8f acc = {};
#pragma unroll 4
    for (int kc = 0; kc < NHH_; kc += 32) {
      const v8bf* a0 = (const v8bf*)(abase + kc + khalf * 8);
      const v8bf* a1 = (const v8bf*)(abase + kc + 16 + khalf * 8);
      const v8bf* bp = (const v8bf*)(bbase + kc + khalf * 8);
      acc = wmma_bf16(join16(a0[0], a1[0]), join16(bp[0], bp[2]), acc);
    }
    const float bias = fc1b[colg];
#pragma unroll
    for (int r = 0; r < 8; ++r)
      hid[m0 + r + khalf * 8][colg] = tanhf(acc[r] + bias);
  }
  __syncthreads();

  if (tid < 2 * B_) {
    int b = tid >> 1, o = tid & 1;
    float s = fc2b[o];
    for (int k = 0; k < H_; ++k) s += hid[b][k] * fc2W[o * H_ + k];
    out[b * 2 + o] = s;
  }
}

// ---------------------------------------------------------------------------
extern "C" void kernel_launch(void* const* d_in, const int* in_sizes, int n_in,
                              void* d_out, int out_size, void* d_ws, size_t ws_size,
                              hipStream_t stream) {
  const int*   src       = (const int*)  d_in[0];
  const int*   input_len = (const int*)  d_in[1];
  const float* fix_src   = (const float*)d_in[2];
  const float* emb_table = (const float*)d_in[3];
  const float* Wi        = (const float*)d_in[4];
  const float* bi        = (const float*)d_in[5];
  const float* Wh        = (const float*)d_in[6];
  const float* bh        = (const float*)d_in[7];
  const float* fc1_W     = (const float*)d_in[8];
  const float* fc1_b     = (const float*)d_in[9];
  const float* fc2_W     = (const float*)d_in[10];
  const float* fc2_b     = (const float*)d_in[11];
  float* out = (float*)d_out;

  // workspace layout (~34.9 MB)
  char* w = (char*)d_ws;
  __bf16* Wi16  = (__bf16*)w;  w += (size_t)H_ * D_   * 2;
  __bf16* Wh16  = (__bf16*)w;  w += (size_t)H_ * NHH_ * 2;
  __bf16* fc116 = (__bf16*)w;  w += (size_t)D_ * NHH_ * 2;
  float*  bvec  = (float*)w;   w += 1024;
  float*  xproj = (float*)w;   w += (size_t)S_ * B_ * H_ * 4;
  __bf16* sel16 = (__bf16*)w;  w += (size_t)B_ * NHH_ * 2;

  convert_weights<<<512, 256, 0, stream>>>(Wi, Wh, fc1_W, bi, bh,
                                           Wi16, Wh16, fc116, bvec);
  xproj_gemm<<<4096, 256, 0, stream>>>(src, emb_table, Wi16, bvec, xproj);
  recurrence<<<4, 512, 0, stream>>>(input_len, fix_src, Wh16, xproj, sel16);
  final_fc<<<1, 512, 0, stream>>>(sel16, fc116, fc1_b, fc2_W, fc2_b, out);
}
